// Decoder_35708358099411
// MI455X (gfx1250) — matrix operand order repaired and checked by reading, not compiled, not run on hardware
//
#include <hip/hip_runtime.h>
#include <math.h>

// ---------------------------------------------------------------------------
// Types for WMMA fragments (gfx1250, wave32)
// ---------------------------------------------------------------------------
typedef __attribute__((ext_vector_type(16))) __bf16 v16bf;
typedef __attribute__((ext_vector_type(8)))  __bf16 v8bf;
typedef __attribute__((ext_vector_type(8)))  float  v8f;

#define SEQLEN 512
#define BATCH  64
#define FEAT   512
#define HID    1024
#define WIN    16
#define OUT_F  512

__device__ __forceinline__ float sigm(float x) { return 1.0f / (1.0f + __expf(-x)); }

__device__ __forceinline__ v16bf load_frag_b(const __bf16* p) {
    v8bf b0 = *(const v8bf*)p;
    v8bf b1 = *(const v8bf*)(p + 8);
    v16bf r;
#pragma unroll
    for (int j = 0; j < 8; ++j) { r[j] = b0[j]; r[j + 8] = b1[j]; }
    return r;
}

__device__ __forceinline__ v16bf load_frag_a(const __bf16* p) {
    // A lane holds K = [kb..kb+16)
    v8bf lo = *(const v8bf*)p;
    v8bf hi = *(const v8bf*)(p + 8);
    v16bf r;
#pragma unroll
    for (int j = 0; j < 8; ++j) { r[j] = lo[j]; r[j + 8] = hi[j]; }
    return r;
}

// Async DMA of 16 bytes global -> LDS (tracked by ASYNCcnt, no VGPR staging).
__device__ __forceinline__ void async_copy16(unsigned lds_off, const void* gptr) {
    asm volatile("global_load_async_to_lds_b128 %0, %1, off"
                 :: "v"(lds_off), "v"(gptr) : "memory");
}
__device__ __forceinline__ void wait_async_le2() {
    asm volatile("s_wait_asynccnt 0x2" ::: "memory");
}
__device__ __forceinline__ void wait_async_0() {
    asm volatile("s_wait_asynccnt 0x0" ::: "memory");
}

// ---------------------------------------------------------------------------
// Setup: f32 -> bf16 conversion
// ---------------------------------------------------------------------------
__global__ void cvt_f32_bf16_kernel(const float* __restrict__ src,
                                    __bf16* __restrict__ dst, int n) {
    int i = blockIdx.x * blockDim.x + threadIdx.x;
    if (i < n) dst[i] = (__bf16)src[i];
}

// WaT[sp][f][h] = Wa[sp][h][f]  (so B-fragments along K=HID are contiguous)
__global__ void transpose_wa_kernel(const float* __restrict__ Wa,
                                    __bf16* __restrict__ WaT) {
    int i = blockIdx.x * blockDim.x + threadIdx.x; // 2*HID*FEAT
    if (i >= 2 * HID * FEAT) return;
    int sp = i / (HID * FEAT);
    int r  = i % (HID * FEAT);
    int h  = r / FEAT, f = r % FEAT;
    WaT[(sp * FEAT + f) * HID + h] = (__bf16)Wa[i];
}

// ---------------------------------------------------------------------------
// Gconst[l] = h0[l] @ Whh_l.T + bih_l + bhh_l   -> [3, 64, 4096] f32 (one time)
// grid: 3*256 blocks (layer, n-tile of 4H), 128 threads (wave = M-tile)
// ---------------------------------------------------------------------------
__global__ void gconst_kernel(const float* __restrict__ h0,
                              const float* __restrict__ Whh0,
                              const float* __restrict__ WhhR,
                              const float* __restrict__ bih0,
                              const float* __restrict__ bhh0,
                              const float* __restrict__ bihR,
                              const float* __restrict__ bhhR,
                              float* __restrict__ G) {
    int blk   = blockIdx.x;
    int layer = blk >> 8;          // / 256
    int ntile = blk & 255;
    int wave  = threadIdx.x >> 5;
    int lane  = threadIdx.x & 31;

    const float* W   = (layer == 0) ? Whh0 : (WhhR + (size_t)(layer - 1) * 4 * HID * HID);
    const float* bih = (layer == 0) ? bih0 : (bihR + (layer - 1) * 4 * HID);
    const float* bhh = (layer == 0) ? bhh0 : (bhhR + (layer - 1) * 4 * HID);
    const float* A   = h0 + (size_t)layer * BATCH * HID;

    int arow = wave * 16 + (lane & 15);
    int kbA  = (lane >> 4) * 16;
    int ncol = ntile * 16 + (lane & 15);
    int kbB  = (lane >> 4) * 16;

    v8f acc = {};
    for (int kt = 0; kt < HID; kt += 32) {
        const float* ap = A + (size_t)arow * HID + kt + kbA;
        v16bf a, b;
#pragma unroll
        for (int j = 0; j < 8; ++j) { a[j] = (__bf16)ap[j]; a[j + 8] = (__bf16)ap[8 + j]; }
        const float* bp = W + (size_t)ncol * HID + kt + kbB;
#pragma unroll
        for (int j = 0; j < 16; ++j) b[j] = (__bf16)bp[j];
        acc = __builtin_amdgcn_wmma_f32_16x16x32_bf16(false, a, false, b, (short)0, acc,
                                                      false, false);
    }
    float bias = bih[ncol] + bhh[ncol];
#pragma unroll
    for (int r = 0; r < 8; ++r) {
        int brow = wave * 16 + r + 8 * (lane >> 4);
        G[((size_t)layer * BATCH + brow) * (4 * HID) + ncol] = acc[r] + bias;
    }
}

// ---------------------------------------------------------------------------
// proj = s_bf16 [64,1024] @ WaT_sp.T  -> proj f32 [64,512]
// grid 32 (n-tiles of FEAT), block 128 (wave = M-tile)
// ---------------------------------------------------------------------------
__global__ void proj_kernel(const __bf16* __restrict__ S,
                            const __bf16* __restrict__ WaT, // [FEAT, HID] this speaker
                            float* __restrict__ proj) {
    int ntile = blockIdx.x;
    int wave = threadIdx.x >> 5, lane = threadIdx.x & 31;
    int arow = wave * 16 + (lane & 15);
    int kbA  = (lane >> 4) * 16;
    int ncol = ntile * 16 + (lane & 15);
    int kbB  = (lane >> 4) * 16;

    v8f acc = {};
    for (int kt = 0; kt < HID; kt += 32) {
        v16bf a = load_frag_a(S + (size_t)arow * HID + kt + kbA);
        v16bf b = load_frag_b(WaT + (size_t)ncol * HID + kt + kbB);
        acc = __builtin_amdgcn_wmma_f32_16x16x32_bf16(false, a, false, b, (short)0, acc,
                                                      false, false);
    }
#pragma unroll
    for (int r = 0; r < 8; ++r) {
        int brow = wave * 16 + r + 8 * (lane >> 4);
        proj[brow * FEAT + ncol] = acc[r];
    }
}

// ---------------------------------------------------------------------------
// Attention for one timestep: e[w]=<proj[b],h[t-W+w,b]>, softmax over 33,
// c_t[b] = sum_w a[w]*h[..]; writes c_t in bf16.  grid 64 (batch), block 256.
// Zero padding: out-of-range rows contribute e=0 (dot with zeros) and no data.
// ---------------------------------------------------------------------------
__global__ void attn_kernel(const float* __restrict__ h,    // [T,B,F]
                            const float* __restrict__ proj, // [B,F]
                            __bf16* __restrict__ Ct,        // [B,F] bf16
                            int t) {
    __shared__ float sh_e[33];
    __shared__ float sh_a[33];
    int b = blockIdx.x;
    int tid = threadIdx.x, wave = tid >> 5, lane = tid & 31;

    for (int w = wave; w < 33; w += 8) {
        int tt = t - WIN + w;
        float sum = 0.0f;
        if (tt >= 0 && tt < SEQLEN) {
            const float* hp = h + ((size_t)tt * BATCH + b) * FEAT;
            const float* pp = proj + (size_t)b * FEAT;
            for (int f = lane; f < FEAT; f += 32) sum += pp[f] * hp[f];
        }
#pragma unroll
        for (int off = 16; off; off >>= 1) sum += __shfl_xor(sum, off, 32);
        if (lane == 0) sh_e[w] = sum; // 0 for padded rows
    }
    __syncthreads();
    if (tid < 32) {
        float m = -1e30f;
        for (int w = lane; w < 33; w += 32) m = fmaxf(m, sh_e[w]);
#pragma unroll
        for (int off = 16; off; off >>= 1) m = fmaxf(m, __shfl_xor(m, off, 32));
        float ssum = 0.0f;
        for (int w = lane; w < 33; w += 32) ssum += __expf(sh_e[w] - m);
#pragma unroll
        for (int off = 16; off; off >>= 1) ssum += __shfl_xor(ssum, off, 32);
        for (int w = lane; w < 33; w += 32) sh_a[w] = __expf(sh_e[w] - m) / ssum;
    }
    __syncthreads();
    int f = tid * 2;
    float c0 = 0.0f, c1 = 0.0f;
    for (int w = 0; w < 33; ++w) {
        int tt = t - WIN + w;
        if (tt < 0 || tt >= SEQLEN) continue;
        const float* hp = h + ((size_t)tt * BATCH + b) * FEAT + f;
        float aw = sh_a[w];
        c0 += aw * hp[0];
        c1 += aw * hp[1];
    }
    Ct[b * FEAT + f]     = (__bf16)c0;
    Ct[b * FEAT + f + 1] = (__bf16)c1;
}

// ---------------------------------------------------------------------------
// Fused LSTM cell with async-LDS weight staging.
//   gates = X @ W.T + Gconst;  x = sig(o)*tanh(sig(f)*c0 + sig(i)*tanh(g))
// grid 64 (n-tiles of HID), block 128 (wave = M-tile, all 4 gate accums in
// registers).  The 4-gate 16x32 weight tile (4KB) is DMA'd into LDS once per
// block per K-step (double-buffered), instead of each wave re-loading it.
// ---------------------------------------------------------------------------
__global__ void lstm_kernel(const __bf16* __restrict__ X,  // [64,K] bf16
                            int K,
                            const __bf16* __restrict__ W,  // [4H,K] bf16 row-major
                            const float*  __restrict__ G,  // [64,4H] const part
                            const float*  __restrict__ c0, // [64,H]
                            __bf16* __restrict__ Xout) {   // [64,H] bf16
    // [buf][gate(4)][row(16)][k(32)] bf16 = 4KB per buffer
    __shared__ __bf16 wbuf[2][4 * 16 * 32];

    int ntile = blockIdx.x;
    int tid  = threadIdx.x;
    int wave = tid >> 5, lane = tid & 31;
    int n0   = ntile * 16;
    int arow = wave * 16 + (lane & 15);
    int kbA  = (lane >> 4) * 16;
    int n    = n0 + (lane & 15);

    // Stage one K-tile of all 4 gate weight blocks: 256 x 16B chunks,
    // 128 threads x 2 async copies each.
#define STAGE(kt_, buf_)                                                        \
    {                                                                           \
        _Pragma("unroll")                                                       \
        for (int i = 0; i < 2; ++i) {                                           \
            int c   = tid + 128 * i;      /* chunk id            */             \
            int g   = c >> 6;             /* 64 chunks per gate  */             \
            int cc  = c & 63;                                                   \
            int row = cc >> 2;            /* 4 chunks per row    */             \
            int ko  = (cc & 3) * 8;       /* bf16 offset in row  */             \
            const __bf16* gp = W + (size_t)(g * HID + n0 + row) * K + (kt_) + ko;\
            unsigned laddr = (unsigned)(size_t)&wbuf[buf_][c * 8];              \
            async_copy16(laddr, gp);                                            \
        }                                                                       \
    }

    STAGE(0, 0);

    v8f ai = {}, af = {}, ag = {}, ao = {};
    const int nk = K / 32;
    for (int ktile = 0; ktile < nk; ++ktile) {
        int kt  = ktile * 32;
        int buf = ktile & 1;
        if (ktile + 1 < nk) {
            STAGE(kt + 32, buf ^ 1);
            wait_async_le2();   // force the 2 older copies (current buf) done
        } else {
            wait_async_0();
        }
        __syncthreads();        // all waves' chunks of `buf` visible

        v16bf a = load_frag_a(X + (size_t)arow * K + kt + kbA);
        const __bf16* base = &wbuf[buf][(lane & 15) * 32 + (lane >> 4) * 16];
        v16bf bi = load_frag_b(base + 0 * 512);
        ai = __builtin_amdgcn_wmma_f32_16x16x32_bf16(false, a, false, bi, (short)0, ai, false, false);
        v16bf bf = load_frag_b(base + 1 * 512);
        af = __builtin_amdgcn_wmma_f32_16x16x32_bf16(false, a, false, bf, (short)0, af, false, false);
        v16bf bg = load_frag_b(base + 2 * 512);
        ag = __builtin_amdgcn_wmma_f32_16x16x32_bf16(false, a, false, bg, (short)0, ag, false, false);
        v16bf bo = load_frag_b(base + 3 * 512);
        ao = __builtin_amdgcn_wmma_f32_16x16x32_bf16(false, a, false, bo, (short)0, ao, false, false);

        __syncthreads();        // don't let anyone re-stage `buf^1` early
    }
#undef STAGE

#pragma unroll
    for (int r = 0; r < 8; ++r) {
        int brow = wave * 16 + r + 8 * (lane >> 4);
        const float* Gb = G + (size_t)brow * (4 * HID);
        float gi = ai[r] + Gb[n];
        float gf = af[r] + Gb[HID + n];
        float gg = ag[r] + Gb[2 * HID + n];
        float go = ao[r] + Gb[3 * HID + n];
        float cn = sigm(gf) * c0[(size_t)brow * HID + n] + sigm(gi) * tanhf(gg);
        Xout[(size_t)brow * HID + n] = (__bf16)(sigm(go) * tanhf(cn));
    }
}

// ---------------------------------------------------------------------------
// out = relu(concat(x3, c_t) @ Wfc.T + bfc)  -> f32 [64,512] at d_out[sp][t]
// grid 32 (n-tiles of OUT), block 128 (wave = M-tile); A is a virtual concat.
// ---------------------------------------------------------------------------
__global__ void fc_kernel(const __bf16* __restrict__ X3, // [64,1024]
                          const __bf16* __restrict__ Ct, // [64,512]
                          const __bf16* __restrict__ Wfc,// [512,1536]
                          const float*  __restrict__ bfc,
                          float* __restrict__ out) {     // [64,512]
    int ntile = blockIdx.x;
    int wave = threadIdx.x >> 5, lane = threadIdx.x & 31;
    int arow = wave * 16 + (lane & 15);
    int kbA  = (lane >> 4) * 16;
    int n    = ntile * 16 + (lane & 15);
    int kbB  = (lane >> 4) * 16;
    const int KT = HID + FEAT; // 1536

    v8f acc = {};
    for (int kt = 0; kt < KT; kt += 32) {
        const __bf16* ap = (kt < HID)
            ? (X3 + (size_t)arow * HID + kt + kbA)
            : (Ct + (size_t)arow * FEAT + (kt - HID) + kbA);
        v16bf a = load_frag_a(ap);
        v16bf b = load_frag_b(Wfc + (size_t)n * KT + kt + kbB);
        acc = __builtin_amdgcn_wmma_f32_16x16x32_bf16(false, a, false, b, (short)0, acc,
                                                      false, false);
    }
    float bias = bfc[n];
#pragma unroll
    for (int r = 0; r < 8; ++r) {
        int brow = wave * 16 + r + 8 * (lane >> 4);
        out[(size_t)brow * OUT_F + n] = fmaxf(acc[r] + bias, 0.0f);
    }
}

// ---------------------------------------------------------------------------
// Host side
// ---------------------------------------------------------------------------
extern "C" void kernel_launch(void* const* d_in, const int* in_sizes, int n_in,
                              void* d_out, int out_size, void* d_ws, size_t ws_size,
                              hipStream_t stream) {
    const float* h    = (const float*)d_in[0];   // [512,64,512]
    const float* Wa   = (const float*)d_in[1];   // [2,1024,512]
    const float* Wih0 = (const float*)d_in[2];   // [4096,512]
    const float* Whh0 = (const float*)d_in[3];   // [4096,1024]
    const float* bih0 = (const float*)d_in[4];
    const float* bhh0 = (const float*)d_in[5];
    const float* WihR = (const float*)d_in[6];   // [2,4096,1024]
    const float* WhhR = (const float*)d_in[7];   // [2,4096,1024]
    const float* bihR = (const float*)d_in[8];
    const float* bhhR = (const float*)d_in[9];
    const float* Wfc  = (const float*)d_in[10];  // [512,1536]
    const float* bfc  = (const float*)d_in[11];
    const float* s0   = (const float*)d_in[12];  // [64,1024]
    const float* h0   = (const float*)d_in[13];  // [3,64,1024]
    const float* c0   = (const float*)d_in[14];  // [3,64,1024]
    float* out        = (float*)d_out;           // [2,512,64,512]

    char* ws = (char*)d_ws;
    size_t off = 0;
    __bf16* WaT   = (__bf16*)(ws + off); off += (size_t)2 * FEAT * HID * 2;           // 2 MB
    __bf16* Wih0b = (__bf16*)(ws + off); off += (size_t)4 * HID * FEAT * 2;           // 4 MB
    __bf16* WihRb = (__bf16*)(ws + off); off += (size_t)2 * 4 * HID * HID * 2;        // 16 MB
    __bf16* Wfcb  = (__bf16*)(ws + off); off += (size_t)OUT_F * (HID + FEAT) * 2;     // 1.5 MB
    float*  G     = (float*) (ws + off); off += (size_t)3 * BATCH * 4 * HID * 4;      // 3 MB
    float*  proj  = (float*) (ws + off); off += (size_t)BATCH * FEAT * 4;
    __bf16* Ct    = (__bf16*)(ws + off); off += (size_t)BATCH * FEAT * 2;
    __bf16* xA    = (__bf16*)(ws + off); off += (size_t)BATCH * HID * 2;
    __bf16* xB    = (__bf16*)(ws + off); off += (size_t)BATCH * HID * 2;
    __bf16* S     = (__bf16*)(ws + off); off += (size_t)BATCH * HID * 2;

    // --- one-time weight preparation -------------------------------------
    {
        int n;
        n = 4 * HID * FEAT;
        cvt_f32_bf16_kernel<<<(n + 255) / 256, 256, 0, stream>>>(Wih0, Wih0b, n);
        n = 2 * 4 * HID * HID;
        cvt_f32_bf16_kernel<<<(n + 255) / 256, 256, 0, stream>>>(WihR, WihRb, n);
        n = OUT_F * (HID + FEAT);
        cvt_f32_bf16_kernel<<<(n + 255) / 256, 256, 0, stream>>>(Wfc, Wfcb, n);
        n = BATCH * HID;
        cvt_f32_bf16_kernel<<<(n + 255) / 256, 256, 0, stream>>>(s0, S, n);
        n = 2 * HID * FEAT;
        transpose_wa_kernel<<<(n + 255) / 256, 256, 0, stream>>>(Wa, WaT);
        gconst_kernel<<<3 * 256, 128, 0, stream>>>(h0, Whh0, WhhR, bih0, bhh0, bihR,
                                                   bhhR, G);
    }

    // --- sequential decode: s carries across timesteps AND speakers ------
    for (int sp = 0; sp < 2; ++sp) {
        const __bf16* WaT_sp = WaT + (size_t)sp * FEAT * HID;
        for (int t = 0; t < SEQLEN; ++t) {
            proj_kernel<<<FEAT / 16, 128, 0, stream>>>(S, WaT_sp, proj);
            attn_kernel<<<BATCH, 256, 0, stream>>>(h, proj, Ct, t);
            lstm_kernel<<<HID / 16, 128, 0, stream>>>(Ct, FEAT, Wih0b,
                                                      G, c0, xA);
            lstm_kernel<<<HID / 16, 128, 0, stream>>>(xA, HID,
                                                      WihRb,
                                                      G + (size_t)1 * BATCH * 4 * HID,
                                                      c0 + (size_t)1 * BATCH * HID, xB);
            lstm_kernel<<<HID / 16, 128, 0, stream>>>(xB, HID,
                                                      WihRb + (size_t)4 * HID * HID,
                                                      G + (size_t)2 * BATCH * 4 * HID,
                                                      c0 + (size_t)2 * BATCH * HID, S);
            fc_kernel<<<OUT_F / 16, 128, 0, stream>>>(S, Ct, Wfcb, bfc,
                out + ((size_t)sp * SEQLEN + t) * BATCH * OUT_F);
        }
    }
}
